// DLRM_Net_5669356831571
// MI455X (gfx1250) — compile-verified, matrix operand order checked by reading, not run on hardware
//
#include <hip/hip_runtime.h>
#include <hip/hip_bf16.h>
#include <stdint.h>

// ---------------------------------------------------------------------------
// DLRM forward, fully WMMA-based for gfx1250 (wave32, 16x16x32 f16 WMMA),
// with async global<->LDS staging (ASYNCcnt) via the gfx1250 async builtins.
//
// Pipeline (all on `stream`):
//   K0 x6 : pad/convert f32 weights -> f16 (WMMA-friendly shapes) in d_ws
//   K1    : bottom MLP  13(->32) -> 512 -> 256 -> 64, fused per 16-row wave
//   K2    : embedding gather (async b128 -> LDS stage) + T·T^T interaction
//   K3    : top MLP 415(->416) -> 512 -> 256 -> 1(->16), fused, f32 out
// ---------------------------------------------------------------------------

typedef __attribute__((ext_vector_type(16))) _Float16 v16h;
typedef __attribute__((ext_vector_type(8)))  float    v8f;
typedef __attribute__((ext_vector_type(4)))  int      v4i;

#define BATCH   4096
#define VOCAB   100000
#define N_EMB   26
#define EMB_DIM 64
#define NI      27          // 1 + N_EMB
#define P_REAL  415         // 64 + 351
#define P_PAD   416

// Async global<->LDS availability probes (compile-safe on both toolchains).
#if defined(__has_builtin)
#  if __has_builtin(__builtin_amdgcn_global_load_async_to_lds_b128) && \
      __has_builtin(__builtin_amdgcn_s_wait_asynccnt)
#    define HAVE_ASYNC_LD 1
#  endif
#  if __has_builtin(__builtin_amdgcn_global_store_async_from_lds_b128) && \
      __has_builtin(__builtin_amdgcn_s_wait_asynccnt)
#    define HAVE_ASYNC_ST 1
#  endif
#endif
#ifndef HAVE_ASYNC_LD
#  define HAVE_ASYNC_LD 0
#endif
#ifndef HAVE_ASYNC_ST
#  define HAVE_ASYNC_ST 0
#endif

#define AS1 __attribute__((address_space(1)))
#define AS3 __attribute__((address_space(3)))

#if HAVE_ASYNC_LD
__device__ __forceinline__ void async_g2l_b128(const void* g, void* l) {
  __builtin_amdgcn_global_load_async_to_lds_b128((AS1 v4i*)g, (AS3 v4i*)l, 0, 0);
}
#endif
#if HAVE_ASYNC_ST
__device__ __forceinline__ void async_l2g_b128(void* g, const void* l) {
  __builtin_amdgcn_global_store_async_from_lds_b128((AS1 v4i*)g, (AS3 v4i*)l, 0, 0);
}
#endif

// Workspace layout (in _Float16 elements)
#define OFF_W0B 0                       // 512 x 32
#define OFF_W1B (OFF_W0B + 512*32)      // 256 x 512
#define OFF_W2B (OFF_W1B + 256*512)     // 64  x 256
#define OFF_W0T (OFF_W2B + 64*256)      // 512 x 416
#define OFF_W1T (OFF_W0T + 512*416)     // 256 x 512
#define OFF_W2T (OFF_W1T + 256*512)     // 16  x 256
#define OFF_X16 (OFF_W2T + 16*256)      // 4096 x 64   (bottom-MLP output, f16)
#define OFF_P   (OFF_X16 + BATCH*64)    // 4096 x 416  (top-MLP input, f16)

// ---------------------------------------------------------------------------
// Fragment load matching CDNA5 16-bit A/B WMMA VGPR layout (ISA 7.12.2):
//   lane l: row r = l&15, k-half kh = l>>4
//   elems 0..7  -> k0 + 8*kh + e          (16B contiguous)
//   elems 8..15 -> k0 + 16 + 8*kh + (e-8) (16B contiguous)
// ---------------------------------------------------------------------------
__device__ __forceinline__ v16h ld_frag(const _Float16* __restrict__ base,
                                        int stride, int k0) {
  const int l  = threadIdx.x & 31;
  const int r  = l & 15;
  const int kh = l >> 4;
  const _Float16* p = base + r * stride + k0 + 8 * kh;
  v16h f;
#pragma unroll
  for (int e = 0; e < 8; ++e) f[e] = p[e];
#pragma unroll
  for (int e = 0; e < 8; ++e) f[8 + e] = p[16 + e];
  return f;
}

// One fused MLP layer for one wave's 16-row tile (A frags in registers,
// B streams from L2-resident global weights).
template <int KP, bool RELU>
__device__ __forceinline__ void mlp_layer(const _Float16* __restrict__ in,
                                          const float* __restrict__ bias,
                                          const _Float16* __restrict__ W,
                                          int Np,
                                          _Float16* __restrict__ out) {
  constexpr int NK = KP / 32;
  v16h afr[NK];
#pragma unroll
  for (int k = 0; k < NK; ++k) afr[k] = ld_frag(in, KP, k * 32);

  const int l  = threadIdx.x & 31;
  const int rb = (l >> 4) * 8;   // D-matrix row base for this lane half

  for (int n0 = 0; n0 < Np; n0 += 16) {
    if (n0 + 16 < Np) __builtin_prefetch(W + (n0 + 16) * KP, 0, 0);
    const _Float16* wb = W + n0 * KP;
    v8f acc = {};
#pragma unroll
    for (int k = 0; k < NK; ++k) {
      v16h bfr = ld_frag(wb, KP, k * 32);
      acc = __builtin_amdgcn_wmma_f32_16x16x32_f16(
          false, afr[k], false, bfr, (short)0, acc, false, false);
    }
    const int n  = n0 + (l & 15);
    const float bv = bias[n];
#pragma unroll
    for (int v = 0; v < 8; ++v) {
      float x = acc[v] + bv;
      if (RELU) x = x > 0.f ? x : 0.f;
      out[(rb + v) * Np + n] = (_Float16)x;
    }
  }
}

// ---------------------------------------------------------------------------
// K0: f32 -> f16 weight convert with zero padding.
// ---------------------------------------------------------------------------
__global__ void k_prep_w(const float* __restrict__ src, _Float16* __restrict__ dst,
                         int rows, int cols, int rows_p, int cols_p) {
  int tot = rows_p * cols_p;
  for (int i = blockIdx.x * blockDim.x + threadIdx.x; i < tot;
       i += gridDim.x * blockDim.x) {
    int r = i / cols_p, c = i % cols_p;
    float v = (r < rows && c < cols) ? src[r * cols + c] : 0.f;
    dst[i] = (_Float16)v;
  }
}

// ---------------------------------------------------------------------------
// K1: bottom MLP. 128 threads = 4 waves, each wave owns 16 batch rows.
// Dynamic LDS: per wave two 16x512 f16 ping-pong buffers (32 KB) -> 128 KB.
// ---------------------------------------------------------------------------
#define BUFH (16 * 512)

__global__ void k_bottom(const float* __restrict__ dx,
                         const _Float16* __restrict__ W0,
                         const float* __restrict__ b0,
                         const _Float16* __restrict__ W1,
                         const float* __restrict__ b1,
                         const _Float16* __restrict__ W2,
                         const float* __restrict__ b2,
                         _Float16* __restrict__ X16) {
  extern __shared__ _Float16 smem[];
  const int w = threadIdx.x >> 5, l = threadIdx.x & 31;
  _Float16* buf0 = smem + w * 2 * BUFH;
  _Float16* buf1 = buf0 + BUFH;
  const int r0 = blockIdx.x * 64 + w * 16;   // wave's batch-row base

#if HAVE_ASYNC_LD
  // Async-copy the contiguous 16x13 f32 block (832 B, 16B-aligned since
  // r0 % 16 == 0) into LDS staging (reuse buf1), then pad/convert into buf0.
  {
    const char* gsrc = (const char*)(dx + (size_t)r0 * 13);
    char*       lst  = (char*)buf1;
    for (int c = l; c < 52; c += 32)            // 52 x 16B chunks
      async_g2l_b128(gsrc + c * 16, lst + c * 16);
    __builtin_amdgcn_s_wait_asynccnt(0);
    const float* stage = (const float*)buf1;
#pragma unroll
    for (int rr = 0; rr < 16; ++rr)
      for (int cc = l; cc < 32; cc += 32) {
        float v = (cc < 13) ? stage[rr * 13 + cc] : 0.f;
        buf0[rr * 32 + cc] = (_Float16)v;
      }
  }
#else
#pragma unroll
  for (int rr = 0; rr < 16; ++rr)
    for (int cc = l; cc < 32; cc += 32) {
      float v = (cc < 13) ? dx[(r0 + rr) * 13 + cc] : 0.f;
      buf0[rr * 32 + cc] = (_Float16)v;
    }
#endif
  __syncthreads();

  mlp_layer<32,  true >(buf0, b0, W0, 512, buf1);
  __syncthreads();
  mlp_layer<512, true >(buf1, b1, W1, 256, buf0);
  __syncthreads();
  mlp_layer<256, false>(buf0, b2, W2, 64,  buf1);
  __syncthreads();

  // Write 16 x 64 bottleneck to global (f16) — contiguous 2 KB per wave.
#if HAVE_ASYNC_ST
  {
    char*       gdst = (char*)(X16 + (size_t)r0 * 64);
    const char* lsrc = (const char*)buf1;
    for (int c = l; c < 128; c += 32)           // 128 x 16B chunks
      async_l2g_b128(gdst + c * 16, lsrc + c * 16);
    __builtin_amdgcn_s_wait_asynccnt(0);
  }
#else
#pragma unroll
  for (int rr = 0; rr < 16; ++rr)
    for (int cc = l; cc < 64; cc += 32)
      X16[(r0 + rr) * 64 + cc] = buf1[rr * 64 + cc];
#endif
}

// ---------------------------------------------------------------------------
// K2: embedding gather + interaction.  4 waves/block, 1 sample per wave.
// Embedding rows (256 B each) are async-DMA'd into an LDS f32 stage, then
// converted to the f16 T tile (32x64, rows 27..31 zero).  Z = T·T^T via
// 2x2 tiles of 16x16x32 WMMA; B fragment of T^T == row-major T.
// Per-wave LDS: 4096 B (T) + 6656 B (stage) = 10752 B; x4 waves = 43008 B.
// ---------------------------------------------------------------------------
#define INT_WAVE_BYTES (32 * 64 * 2 + N_EMB * 64 * 4)

__global__ void k_interact(const int* __restrict__ sidx,
                           const float* __restrict__ emb,
                           const _Float16* __restrict__ X16,
                           _Float16* __restrict__ P) {
  extern __shared__ char smemc[];
  const int w = threadIdx.x >> 5, l = threadIdx.x & 31;
  char* wbase = smemc + w * INT_WAVE_BYTES;
  _Float16* T     = (_Float16*)wbase;
  float*    stage = (float*)(wbase + 32 * 64 * 2);
  const int b = blockIdx.x * 4 + w;          // sample

  // Row 0: bottom-MLP output x[b] (already f16).
  T[2 * l]     = X16[b * 64 + 2 * l];
  T[2 * l + 1] = X16[b * 64 + 2 * l + 1];

#if HAVE_ASYNC_LD
  // 26 rows x 16 chunks of 16 B, lane-strided async gather into LDS stage.
#pragma unroll 1
  for (int c = l; c < N_EMB * 16; c += 32) {
    int j = c >> 4, o = c & 15;
    int idx = sidx[j * BATCH + b];
    const float* er = emb + ((size_t)j * VOCAB + (size_t)idx) * EMB_DIM;
    async_g2l_b128(er + o * 4, stage + j * 64 + o * 4);
  }
  __builtin_amdgcn_s_wait_asynccnt(0);
  // Convert staged f32 -> f16 T rows 1..26.
#pragma unroll 1
  for (int i = l; i < N_EMB * 64; i += 32)
    T[64 + i] = (_Float16)stage[i];
#else
#pragma unroll 1
  for (int j = 0; j < N_EMB; ++j) {
    int idx = sidx[j * BATCH + b];
    const float* er = emb + ((size_t)j * VOCAB + (size_t)idx) * EMB_DIM;
    T[(1 + j) * 64 + 2 * l]     = (_Float16)er[2 * l];
    T[(1 + j) * 64 + 2 * l + 1] = (_Float16)er[2 * l + 1];
  }
#endif
  // Rows 27..31: zero pad.
  for (int i = l; i < 5 * 64; i += 32) T[27 * 64 + i] = (_Float16)0.f;
  __syncthreads();

  _Float16* pb = P + (size_t)b * P_PAD;

  // 2x2 WMMA tiles of Z = T·T^T; scatter strict lower triangle.
#pragma unroll
  for (int mi = 0; mi < 32; mi += 16) {
#pragma unroll
    for (int ni = 0; ni < 32; ni += 16) {
      v8f acc = {};
#pragma unroll
      for (int k0 = 0; k0 < 64; k0 += 32) {
        v16h a  = ld_frag(T + mi * 64, 64, k0);
        v16h bb = ld_frag(T + ni * 64, 64, k0);
        acc = __builtin_amdgcn_wmma_f32_16x16x32_f16(
            false, a, false, bb, (short)0, acc, false, false);
      }
      const int n  = ni + (l & 15);
      const int rb = mi + (l >> 4) * 8;
#pragma unroll
      for (int v = 0; v < 8; ++v) {
        int m = rb + v;
        if (m < NI && n < NI && m > n)
          pb[64 + (m * (m - 1)) / 2 + n] = (_Float16)acc[v];
      }
    }
  }

  // p[b][0..63] = x[b];  p[b][415] = 0 (pad column; W0t pad col is zero too).
  pb[2 * l]     = T[2 * l];
  pb[2 * l + 1] = T[2 * l + 1];
  if (l == 0) pb[P_REAL] = (_Float16)0.f;
}

// ---------------------------------------------------------------------------
// K3: top MLP.  The 16x416 f16 p tile (13312 B, contiguous & 16B aligned)
// is async-DMA'd straight into LDS; final layer N padded 1 -> 16, lanes with
// n==0 hold the real column and write f32 output directly.
// ---------------------------------------------------------------------------
__global__ void k_top(const _Float16* __restrict__ P,
                      const _Float16* __restrict__ W0,
                      const float* __restrict__ b0,
                      const _Float16* __restrict__ W1,
                      const float* __restrict__ b1,
                      const _Float16* __restrict__ W2,
                      const float* __restrict__ b2,
                      float* __restrict__ out) {
  extern __shared__ _Float16 smem[];
  const int w = threadIdx.x >> 5, l = threadIdx.x & 31;
  _Float16* buf0 = smem + w * 2 * BUFH;
  _Float16* buf1 = buf0 + BUFH;
  const int r0 = blockIdx.x * 64 + w * 16;

#if HAVE_ASYNC_LD
  {
    const char* gsrc = (const char*)(P + (size_t)r0 * P_PAD);
    char*       lst  = (char*)buf0;
    for (int c = l; c < 832; c += 32)           // 832 x 16B chunks = 13312 B
      async_g2l_b128(gsrc + c * 16, lst + c * 16);
    __builtin_amdgcn_s_wait_asynccnt(0);
  }
#else
#pragma unroll 1
  for (int rr = 0; rr < 16; ++rr)
    for (int cc = l; cc < P_PAD; cc += 32)
      buf0[rr * P_PAD + cc] = P[(size_t)(r0 + rr) * P_PAD + cc];
#endif
  __syncthreads();

  mlp_layer<416, true>(buf0, b0, W0, 512, buf1);
  __syncthreads();
  mlp_layer<512, true>(buf1, b1, W1, 256, buf0);
  __syncthreads();

  // Final layer: 16x16 single tile, K=256.
  v8f acc = {};
#pragma unroll
  for (int k0 = 0; k0 < 256; k0 += 32) {
    v16h a  = ld_frag(buf0, 256, k0);
    v16h bb = ld_frag(W2,   256, k0);
    acc = __builtin_amdgcn_wmma_f32_16x16x32_f16(
        false, a, false, bb, (short)0, acc, false, false);
  }
  if ((l & 15) == 0) {
    const float bv = b2[0];
    const int rb = (l >> 4) * 8;
#pragma unroll
    for (int v = 0; v < 8; ++v) out[r0 + rb + v] = acc[v] + bv;
  }
}

// ---------------------------------------------------------------------------
extern "C" void kernel_launch(void* const* d_in, const int* in_sizes, int n_in,
                              void* d_out, int out_size, void* d_ws, size_t ws_size,
                              hipStream_t stream) {
  (void)in_sizes; (void)n_in; (void)out_size; (void)ws_size;
  const float* dense_x = (const float*)d_in[0];
  const int*   sidx    = (const int*)d_in[1];
  const float* emb     = (const float*)d_in[2];
  const float* bw0 = (const float*)d_in[3];  const float* bb0 = (const float*)d_in[4];
  const float* bw1 = (const float*)d_in[5];  const float* bb1 = (const float*)d_in[6];
  const float* bw2 = (const float*)d_in[7];  const float* bb2 = (const float*)d_in[8];
  const float* tw0 = (const float*)d_in[9];  const float* tb0 = (const float*)d_in[10];
  const float* tw1 = (const float*)d_in[11]; const float* tb1 = (const float*)d_in[12];
  const float* tw2 = (const float*)d_in[13]; const float* tb2 = (const float*)d_in[14];
  float* out = (float*)d_out;

  _Float16* ws = (_Float16*)d_ws;
  _Float16* W0b = ws + OFF_W0B;  _Float16* W1b = ws + OFF_W1B;
  _Float16* W2b = ws + OFF_W2B;  _Float16* W0t = ws + OFF_W0T;
  _Float16* W1t = ws + OFF_W1T;  _Float16* W2t = ws + OFF_W2T;
  _Float16* X16 = ws + OFF_X16;  _Float16* P   = ws + OFF_P;

  auto gp = [](int tot) { return dim3((tot + 255) / 256); };
  k_prep_w<<<gp(512*32),  256, 0, stream>>>(bw0, W0b, 512, 13,  512, 32);
  k_prep_w<<<gp(256*512), 256, 0, stream>>>(bw1, W1b, 256, 512, 256, 512);
  k_prep_w<<<gp(64*256),  256, 0, stream>>>(bw2, W2b, 64,  256, 64,  256);
  k_prep_w<<<gp(512*416), 256, 0, stream>>>(tw0, W0t, 512, 415, 512, 416);
  k_prep_w<<<gp(256*512), 256, 0, stream>>>(tw1, W1t, 256, 512, 256, 512);
  k_prep_w<<<gp(16*256),  256, 0, stream>>>(tw2, W2t, 1,   256, 16,  256);

  const size_t mlp_lds = 4 * 2 * (size_t)BUFH * sizeof(_Float16);   // 128 KB
  k_bottom<<<BATCH / 64, 128, mlp_lds, stream>>>(
      dense_x, W0b, bb0, W1b, bb1, W2b, bb2, X16);

  const size_t int_lds = 4 * (size_t)INT_WAVE_BYTES;                // 42 KB
  k_interact<<<BATCH / 4, 128, int_lds, stream>>>(sidx, emb, X16, P);

  k_top<<<BATCH / 64, 128, mlp_lds, stream>>>(
      P, W0t, tb0, W1t, tb1, W2t, tb2, out);
}
